// GQAAttention_4389456576826
// MI455X (gfx1250) — matrix operand order repaired and checked by reading, not compiled, not run on hardware
//
#include <hip/hip_runtime.h>
#include <hip/hip_bf16.h>

// ---------------------------------------------------------------------------
// Types for CDNA5 WMMA (wave32): 16x16x32 bf16 -> f32
// ---------------------------------------------------------------------------
typedef __attribute__((ext_vector_type(16))) __bf16 v16bf;
typedef __attribute__((ext_vector_type(8)))  __bf16 v8bf;
typedef __attribute__((ext_vector_type(8)))  float  v8f;

union FragB {
  v16bf v;
  v8bf  h[2];
};

__device__ __forceinline__ v8f wmma_bf16(const FragB& a, const FragB& b, v8f c) {
  // (neg_a, A, neg_b, B, c_mod, C, reuse_a, reuse_b)
  return __builtin_amdgcn_wmma_f32_16x16x32_bf16(false, a.v, false, b.v,
                                                 (short)0, c, false, false);
}

// Async DMA: global -> LDS, 16 bytes per lane, tracked by ASYNCcnt.
// VDST = per-lane LDS byte offset (low 32 bits of a generic LDS pointer),
// VADDR = per-lane 64-bit global address, GV mode (saddr = off).
__device__ __forceinline__ void async_cp_b128(unsigned lds_byte_off, const void* g) {
  asm volatile("global_load_async_to_lds_b128 %0, %1, off"
               :: "v"(lds_byte_off), "v"((unsigned long long)(uintptr_t)g)
               : "memory");
}
__device__ __forceinline__ void wait_async0() {
  asm volatile("s_wait_asynccnt 0x0" ::: "memory");
}

#define T_LEN   2048
#define DIM_    2048
#define NH      32
#define NKVH    8
#define HD      64
#define RANK_   16

// ---------------------------------------------------------------------------
// f32 -> bf16 conversion
// ---------------------------------------------------------------------------
__global__ __launch_bounds__(256) void cvt_bf16_kernel(const float* __restrict__ in,
                                                       __bf16* __restrict__ out, int n) {
  int i = blockIdx.x * 256 + threadIdx.x;
  if (i < n) out[i] = (__bf16)in[i];
}

// ---------------------------------------------------------------------------
// Wq_eff = Wq + lora_B @ lora_A  (rank 16), output bf16
// ---------------------------------------------------------------------------
__global__ __launch_bounds__(256) void wq_eff_kernel(const float* __restrict__ Wq,
                                                     const float* __restrict__ A,
                                                     const float* __restrict__ Bm,
                                                     __bf16* __restrict__ out) {
  int i = blockIdx.x * 256 + threadIdx.x;       // i = row*DIM + col
  int col = i & (DIM_ - 1);
  int row = i >> 11;
  float acc = Wq[i];
#pragma unroll
  for (int r = 0; r < RANK_; r++)
    acc += Bm[row * RANK_ + r] * A[r * DIM_ + col];
  out[i] = (__bf16)acc;
}

// ---------------------------------------------------------------------------
// GEMM: C[M,N] = A[M,K] * W[N,K]^T, bf16 in, f32 accumulate.
// Block: 256 threads = 8 waves (2x4), block tile 64(M) x 128(N),
// wave tile 32x32 = 2x2 WMMA tiles, K step 32.
// A/B tiles staged in LDS via global_load_async_to_lds_b128, double-buffered:
//   iter i: s_wait_asynccnt 0 ; barrier ; issue async for i+1 ; ds_load + wmma
// MODE 0: f32 row-major out; MODE 1: bf16 row-major; MODE 2: bf16 transposed
// (C^T, leading dim ldt) — used to produce V^T for attention.
// ---------------------------------------------------------------------------
template <int MODE>
__global__ __launch_bounds__(256) void gemm_bf16_kernel(const __bf16* __restrict__ A,
                                                        const __bf16* __restrict__ W,
                                                        void* __restrict__ Cout,
                                                        int M, int N, int K, int ldt) {
  __shared__ __bf16 Abuf[2][64][32];    // 8 KB
  __shared__ __bf16 Bbuf[2][128][32];   // 16 KB

  const int tid  = threadIdx.x;
  const int lane = tid & 31;
  const int w    = tid >> 5;
  const int wm   = w >> 2;          // 0..1
  const int wn   = w & 3;           // 0..3
  const int r    = lane & 15;
  const int hi   = lane >> 4;

  const int bm = blockIdx.y * 64;
  const int bn = blockIdx.x * 128;

  // cooperative tile-load assignment: 16B per lane per chunk
  const int arow  = tid >> 2;          // 0..63
  const int acol  = (tid & 3) * 8;     // 0,8,16,24
  const int brow1 = 64 + arow;

  const __bf16* Ablk = A + (size_t)bm * K;
  const __bf16* Wblk = W + (size_t)bn * K;

  v8f acc[2][2] = {};

  // prologue: fill buffer 0
  {
    async_cp_b128((unsigned)(uintptr_t)&Abuf[0][arow][acol],
                  Ablk + (size_t)arow * K + acol);
    async_cp_b128((unsigned)(uintptr_t)&Bbuf[0][arow][acol],
                  Wblk + (size_t)arow * K + acol);
    async_cp_b128((unsigned)(uintptr_t)&Bbuf[0][brow1][acol],
                  Wblk + (size_t)brow1 * K + acol);
  }

  const int nsteps = K >> 5;
  for (int i = 0; i < nsteps; i++) {
    const int buf = i & 1;
    wait_async0();        // this wave's async writes for `buf` have landed
    __syncthreads();      // everyone's writes visible; prev reads of buf^1 done

    if (i + 1 < nsteps) {
      const int k = (i + 1) << 5;
      const int nb = buf ^ 1;
      async_cp_b128((unsigned)(uintptr_t)&Abuf[nb][arow][acol],
                    Ablk + (size_t)arow * K + k + acol);
      async_cp_b128((unsigned)(uintptr_t)&Bbuf[nb][arow][acol],
                    Wblk + (size_t)arow * K + k + acol);
      async_cp_b128((unsigned)(uintptr_t)&Bbuf[nb][brow1][acol],
                    Wblk + (size_t)brow1 * K + k + acol);
    }

    // A fragments: lane row, K chunks [hi*8,+8) and [hi*8+16,+8)
    FragB a0, a1, b0, b1;
    {
      const v8bf* p = (const v8bf*)&Abuf[buf][wm * 32 + r][hi * 8];
      a0.h[0] = p[0]; a0.h[1] = p[2];
      const v8bf* q = (const v8bf*)&Abuf[buf][wm * 32 + 16 + r][hi * 8];
      a1.h[0] = q[0]; a1.h[1] = q[2];
    }
    // B fragments: lane column = W row, K chunks [hi*8,+8) and [hi*8+16,+8)
    {
      const v8bf* p = (const v8bf*)&Bbuf[buf][wn * 32 + r][hi * 8];
      b0.h[0] = p[0]; b0.h[1] = p[2];
      const v8bf* q = (const v8bf*)&Bbuf[buf][wn * 32 + 16 + r][hi * 8];
      b1.h[0] = q[0]; b1.h[1] = q[2];
    }
    acc[0][0] = wmma_bf16(a0, b0, acc[0][0]);
    acc[0][1] = wmma_bf16(a0, b1, acc[0][1]);
    acc[1][0] = wmma_bf16(a1, b0, acc[1][0]);
    acc[1][1] = wmma_bf16(a1, b1, acc[1][1]);
  }

#pragma unroll
  for (int tm = 0; tm < 2; tm++) {
    const int mbase = bm + wm * 32 + tm * 16 + hi * 8;   // rows mbase..mbase+7
#pragma unroll
    for (int tn = 0; tn < 2; tn++) {
      const int n = bn + wn * 32 + tn * 16 + r;
      if (MODE == 0) {
        float* C = (float*)Cout;
#pragma unroll
        for (int v = 0; v < 8; v++)
          C[(size_t)(mbase + v) * N + n] = acc[tm][tn][v];
      } else if (MODE == 1) {
        __bf16* C = (__bf16*)Cout;
#pragma unroll
        for (int v = 0; v < 8; v++)
          C[(size_t)(mbase + v) * N + n] = (__bf16)acc[tm][tn][v];
      } else {
        __bf16* C = (__bf16*)Cout;
        v8bf pk;
#pragma unroll
        for (int v = 0; v < 8; v++) pk[v] = (__bf16)acc[tm][tn][v];
        *(v8bf*)(C + (size_t)n * ldt + mbase) = pk;   // contiguous 16B store
      }
    }
  }
}

// ---------------------------------------------------------------------------
// RoPE: in [T, nHeads*64] f32 row-major -> out [nHeads][T][64] bf16,
// optional scale (1/sqrt(HD) folded into Q).
// ---------------------------------------------------------------------------
__global__ __launch_bounds__(256) void rope_kernel(const float* __restrict__ in,
                                                   __bf16* __restrict__ out,
                                                   int nHeads, float scl) {
  int idx = blockIdx.x * 256 + threadIdx.x;     // t*(nHeads*32) + h*32 + d
  int d = idx & 31;
  int h = (idx >> 5) % nHeads;
  int t = idx / (32 * nHeads);
  if (t >= T_LEN) return;
  const float* row = in + (size_t)t * nHeads * HD + h * HD;
  float x0 = row[d];
  float x1 = row[d + 32];
  float theta = __powf(10000.0f, -(float)d * (1.0f / 32.0f));
  float s, c;
  __sincosf((float)t * theta, &s, &c);
  __bf16* o = out + ((size_t)h * T_LEN + t) * HD;
  o[d]      = (__bf16)((x0 * c - x1 * s) * scl);
  o[d + 32] = (__bf16)((x1 * c + x0 * s) * scl);
}

// ---------------------------------------------------------------------------
// Flash attention (causal, GQA 4:1). Q:[32][T][64], K:[8][T][64], V^T:[8][64][T]
// bf16; out bf16 [T, 32*64]. Block = 8 waves, each wave owns 16 query rows
// (no block barriers; per-wave LDS relay for P).
// ---------------------------------------------------------------------------
__global__ __launch_bounds__(256) void attn_kernel(const __bf16* __restrict__ Q,
                                                   const __bf16* __restrict__ Kt,
                                                   const __bf16* __restrict__ Vt,
                                                   __bf16* __restrict__ O) {
  __shared__ __bf16 ldsP[8][16][128];   // per-wave P tile (C-layout -> A-layout)

  const int lane = threadIdx.x & 31;
  const int w    = threadIdx.x >> 5;
  const int r    = lane & 15;
  const int hi   = lane >> 4;
  const int head = blockIdx.y;
  const int kvh  = head >> 2;                    // 4 q-heads per kv-head
  const int qw   = blockIdx.x * 128 + w * 16;    // this wave's 16 query rows

  const __bf16* Qh = Q  + (size_t)head * T_LEN * HD;
  const __bf16* Kh = Kt + (size_t)kvh  * T_LEN * HD;
  const __bf16* Vh = Vt + (size_t)kvh  * HD * T_LEN;

  // Q A-fragments (constant for the wave): 16 rows x 64 dims = 2 frags
  FragB qf[2];
#pragma unroll
  for (int c = 0; c < 2; c++) {
    const v8bf* p = (const v8bf*)(Qh + (size_t)(qw + r) * HD + c * 32 + hi * 8);
    qf[c].h[0] = p[0];
    qf[c].h[1] = p[2];
  }

  float mrow[8], ssum[8];
  v8f o_acc[4] = {};
#pragma unroll
  for (int v = 0; v < 8; v++) { mrow[v] = -1e30f; ssum[v] = 0.0f; }

  for (int kb = 0; kb <= qw + 15; kb += 128) {
    // speculative prefetch of next key block (K rows + V^T rows)
    if (kb + 128 <= qw + 15) {
      __builtin_prefetch(Kh + (size_t)(kb + 128 + lane * 4) * HD, 0, 1);
      __builtin_prefetch(Vh + (size_t)(lane * 2) * T_LEN + kb + 128, 0, 1);
    }

    v8f S[8];
    float bmax[8];
#pragma unroll
    for (int v = 0; v < 8; v++) bmax[v] = -1e30f;

#pragma unroll
    for (int kt = 0; kt < 8; kt++) {
      const int key = kb + kt * 16 + r;          // this lane's key column
      FragB kf0, kf1;
      const v8bf* p0 = (const v8bf*)(Kh + (size_t)key * HD + hi * 8);
      kf0.h[0] = p0[0]; kf0.h[1] = p0[2];
      const v8bf* p1 = (const v8bf*)(Kh + (size_t)key * HD + 32 + hi * 8);
      kf1.h[0] = p1[0]; kf1.h[1] = p1[2];
      v8f s = {};
      s = wmma_bf16(qf[0], kf0, s);
      s = wmma_bf16(qf[1], kf1, s);
#pragma unroll
      for (int v = 0; v < 8; v++) {
        const int qrow = qw + hi * 8 + v;
        float val = (key > qrow) ? -1e30f : s[v];   // causal mask, branchless
        s[v] = val;
        bmax[v] = fmaxf(bmax[v], val);
      }
      S[kt] = s;
    }

    // row max across the 16 lanes of each half-wave
#pragma unroll
    for (int v = 0; v < 8; v++) {
      float x = bmax[v];
      x = fmaxf(x, __shfl_xor(x, 1, 32));
      x = fmaxf(x, __shfl_xor(x, 2, 32));
      x = fmaxf(x, __shfl_xor(x, 4, 32));
      x = fmaxf(x, __shfl_xor(x, 8, 32));
      bmax[v] = x;
    }

    float scale[8];
#pragma unroll
    for (int v = 0; v < 8; v++) {
      float mn = fmaxf(mrow[v], bmax[v]);
      scale[v] = __expf(mrow[v] - mn);
      mrow[v] = mn;
    }

    // P = exp(S - m): stage into LDS (C-layout scatter), keep partial row sums
    float ps[8];
#pragma unroll
    for (int v = 0; v < 8; v++) ps[v] = 0.0f;
#pragma unroll
    for (int kt = 0; kt < 8; kt++) {
#pragma unroll
      for (int v = 0; v < 8; v++) {
        float p = __expf(S[kt][v] - mrow[v]);
        ps[v] += p;
        ldsP[w][hi * 8 + v][kt * 16 + r] = (__bf16)p;
      }
    }
#pragma unroll
    for (int v = 0; v < 8; v++) ssum[v] = ssum[v] * scale[v] + ps[v];
#pragma unroll
    for (int od = 0; od < 4; od++)
#pragma unroll
      for (int v = 0; v < 8; v++) o_acc[od][v] *= scale[v];

    // cross-lane RAW through LDS within this wave
    asm volatile("s_wait_dscnt 0" ::: "memory");

    // re-read P in A-fragment layout (16 x 128 -> 4 frags)
    FragB pf[4];
#pragma unroll
    for (int c = 0; c < 4; c++) {
      pf[c].h[0] = *(const v8bf*)&ldsP[w][r][c * 32 + hi * 8];
      pf[c].h[1] = *(const v8bf*)&ldsP[w][r][c * 32 + hi * 8 + 16];
    }

    // O += P (16x128) * V (128x64); V^T rows are B columns -> contiguous loads
#pragma unroll
    for (int od = 0; od < 4; od++) {
      v8f acc = o_acc[od];
#pragma unroll
      for (int c = 0; c < 4; c++) {
        FragB vf;
        const v8bf* vp = (const v8bf*)(Vh + (size_t)(od * 16 + r) * T_LEN +
                                       kb + c * 32 + hi * 8);
        vf.h[0] = vp[0]; vf.h[1] = vp[2];
        acc = wmma_bf16(pf[c], vf, acc);
      }
      o_acc[od] = acc;
    }
  }

  // final normalization: reduce row sums across lanes, divide, store bf16
#pragma unroll
  for (int v = 0; v < 8; v++) {
    float x = ssum[v];
    x += __shfl_xor(x, 1, 32);
    x += __shfl_xor(x, 2, 32);
    x += __shfl_xor(x, 4, 32);
    x += __shfl_xor(x, 8, 32);
    ssum[v] = 1.0f / x;
  }
#pragma unroll
  for (int od = 0; od < 4; od++)
#pragma unroll
    for (int v = 0; v < 8; v++) {
      const int row = qw + hi * 8 + v;
      O[(size_t)row * (NH * HD) + head * HD + od * 16 + r] =
          (__bf16)(o_acc[od][v] * ssum[v]);
    }
}

// ---------------------------------------------------------------------------
// Host-side orchestration
// ---------------------------------------------------------------------------
extern "C" void kernel_launch(void* const* d_in, const int* in_sizes, int n_in,
                              void* d_out, int out_size, void* d_ws, size_t ws_size,
                              hipStream_t stream) {
  (void)in_sizes; (void)n_in; (void)out_size; (void)ws_size;

  const float* x  = (const float*)d_in[0];
  const float* Wq = (const float*)d_in[1];
  const float* lA = (const float*)d_in[2];
  const float* lB = (const float*)d_in[3];
  const float* Wk = (const float*)d_in[4];
  const float* Wv = (const float*)d_in[5];
  const float* Wo = (const float*)d_in[6];
  float* out = (float*)d_out;

  char* ws = (char*)d_ws;
  const size_t MB = 1u << 20;
  __bf16* x_bf  = (__bf16*)(ws + 0 * MB);   // 8 MB  [2048,2048]
  __bf16* wq_bf = (__bf16*)(ws + 8 * MB);   // 8 MB  Wq + B*A
  __bf16* wk_bf = (__bf16*)(ws + 16 * MB);  // 2 MB
  __bf16* wv_bf = (__bf16*)(ws + 18 * MB);  // 2 MB
  __bf16* wo_bf = (__bf16*)(ws + 20 * MB);  // 8 MB
  float*  qf32  = (float*) (ws + 28 * MB);  // 16 MB [2048, 2048]
  float*  kf32  = (float*) (ws + 44 * MB);  // 4 MB  [2048, 512]
  __bf16* q_r   = (__bf16*)(ws + 48 * MB);  // 8 MB  [32][2048][64] (rope, *1/8)
  __bf16* k_r   = (__bf16*)(ws + 56 * MB);  // 2 MB  [8][2048][64]  (rope)
  __bf16* v_t   = (__bf16*)(ws + 58 * MB);  // 2 MB  [8][64][2048]  (transposed)
  __bf16* ao_bf = (__bf16*)(ws + 60 * MB);  // 8 MB  attn out [2048, 2048]

  // 1) precision conversion + LoRA fold
  cvt_bf16_kernel<<<(DIM_ * T_LEN) / 256, 256, 0, stream>>>(x, x_bf, DIM_ * T_LEN);
  cvt_bf16_kernel<<<(NKVH * HD * DIM_) / 256, 256, 0, stream>>>(Wk, wk_bf, NKVH * HD * DIM_);
  cvt_bf16_kernel<<<(NKVH * HD * DIM_) / 256, 256, 0, stream>>>(Wv, wv_bf, NKVH * HD * DIM_);
  cvt_bf16_kernel<<<(DIM_ * NH * HD) / 256, 256, 0, stream>>>(Wo, wo_bf, DIM_ * NH * HD);
  wq_eff_kernel<<<(DIM_ * NH * HD) / 256, 256, 0, stream>>>(Wq, lA, lB, wq_bf);

  // 2) projections (async-LDS double-buffered WMMA GEMMs).
  //    V written transposed for the PV B-fragments.
  gemm_bf16_kernel<0><<<dim3(16, 32), 256, 0, stream>>>(x_bf, wq_bf, qf32,
                                                        T_LEN, NH * HD, DIM_, 0);
  gemm_bf16_kernel<0><<<dim3(4, 32), 256, 0, stream>>>(x_bf, wk_bf, kf32,
                                                       T_LEN, NKVH * HD, DIM_, 0);
  gemm_bf16_kernel<2><<<dim3(4, 32), 256, 0, stream>>>(x_bf, wv_bf, v_t,
                                                       T_LEN, NKVH * HD, DIM_, T_LEN);

  // 3) RoPE (softmax scale 1/sqrt(64)=0.125 folded into Q)
  rope_kernel<<<(T_LEN * NH * 32) / 256, 256, 0, stream>>>(qf32, q_r, NH, 0.125f);
  rope_kernel<<<(T_LEN * NKVH * 32) / 256, 256, 0, stream>>>(kf32, k_r, NKVH, 1.0f);

  // 4) causal flash attention (GQA)
  attn_kernel<<<dim3(T_LEN / 128, NH), 256, 0, stream>>>(q_r, k_r, v_t, ao_bf);

  // 5) output projection -> f32
  gemm_bf16_kernel<0><<<dim3(16, 32), 256, 0, stream>>>(ao_bf, wo_bf, out,
                                                        T_LEN, DIM_, NH * HD, 0);
}
